// AttentionMask_43319040147646
// MI455X (gfx1250) — 8-bit matrix operand order repaired and checked by reading, compiled, not run on hardware
//
#include <hip/hip_runtime.h>
#include <math.h>


typedef int v8i __attribute__((ext_vector_type(8)));

__device__ __forceinline__ int a_dw(int v, int h) {
  return ((v >> 1) << 2) + (v & 1) + (h << 1);
}
__device__ __forceinline__ int b_dw(int v, int h) {
  return ((v >> 1) << 2) + (v & 1) + (h << 1);
}

template <int NB>
__device__ __forceinline__ void seg_tile(
    const int* __restrict__ qseg, const int* __restrict__ kseg,
    float* __restrict__ out, float* __restrict__ buf  ,
    const v8i& cpred, int b0, int Q, int K, int q0, int k0,
    int lane, int h, int ln)
{
#pragma unroll
  for (int bb = 0; bb < NB; ++bb) {
    const int b = b0 + bb;

    const int sq  = qseg[b * Q + q0 + ln];
    const int sqd = sq >> 2;
    const int sqv = 1 << ((sq & 3) << 3);
    v8i a;
#pragma unroll
    for (int v = 0; v < 8; ++v) a[v] = (sqd == a_dw(v, h)) ? sqv : 0;

    const int sk  = kseg[b * K + k0 + ln];
    const int skd = sk >> 2;
    const int skv = 1 << ((sk & 3) << 3);
    v8i bm;
#pragma unroll
    for (int v = 0; v < 8; ++v) bm[v] = (skd == b_dw(v, h)) ? skv : 0;

    v8i d = __builtin_amdgcn_wmma_i32_16x16x64_iu8(false, a, false, bm,
                                                   cpred, false, false);

    float* p = buf + bb * 256;
#pragma unroll
    for (int r = 0; r < 8; ++r)
      p[((r + 8 * h) << 4) + ln] = (d[r] == 1) ? 1.0f : 0.0f;
  }

  asm volatile("s_wait_dscnt 0x0" ::: "memory");

  const int row = lane >> 1;
  const int hh  = (lane & 1) << 3;
#pragma unroll
  for (int bb = 0; bb < NB; ++bb) {
    const float4 lo = *(const float4*)(buf + bb * 256 + (row << 4) + hh);
    const float4 hi = *(const float4*)(buf + bb * 256 + (row << 4) + hh + 4);
    float* p = out + (size_t)(b0 + bb) * Q * K + (size_t)(q0 + row) * K
                   + k0 + hh;
    *(float4*)(p)     = lo;
    *(float4*)(p + 4) = hi;
  }
}

__global__ void __launch_bounds__(256)
attn_mask_kernel(const int* __restrict__ expl,
                 const int* __restrict__ qseg,
                 const int* __restrict__ kseg,
                 const int* __restrict__ offp,
                 const int* __restrict__ winp,
                 float* __restrict__ out,
                 int B, int Q, int K, int tilesK, int totalTiles)
{
  __shared__ __align__(16) float xpose[8][4 * 256];

  const int tile = blockIdx.x * 8 + (threadIdx.x >> 5);
  if (tile >= totalTiles) return;

  const int offset = offp[0];
  const int window = winp[0];

  const int w    = threadIdx.x >> 5;
  const int lane = threadIdx.x & 31;
  const int h    = lane >> 4;
  const int ln   = lane & 15;

  const int qt = tile / tilesK;
  const int kt = tile - qt * tilesK;
  const int q0 = qt << 4;
  const int k0 = kt << 4;

  const int lowb = (offset >= 0) ? 0 : -offset;
  const int dmax = q0 + 15 - k0;
  const int dmin = q0 - (k0 + 15);

  if (dmax < lowb || dmin >= window) {
    const int r = lane >> 2;
    const int c = (lane & 3) << 2;
    const float4 z = make_float4(0.f, 0.f, 0.f, 0.f);
    for (int b = 0; b < B; ++b) {
      size_t base = (size_t)b * Q * K;
      *(float4*)(out + base + (size_t)(q0 + r)     * K + (k0 + c)) = z;
      *(float4*)(out + base + (size_t)(q0 + r + 8) * K + (k0 + c)) = z;
    }
    return;
  }

  const int kcol = k0 + ln;
  v8i cpred;
#pragma unroll
  for (int r = 0; r < 8; ++r) {
    const int q    = q0 + r + 8 * h;
    const int diff = q - kcol;
    const bool allow = (diff >= -offset) && (diff >= 0) && (diff < window)
                       && (expl[(size_t)q * K + kcol] != 0);
    cpred[r] = allow ? 0 : -1;
  }

  float* buf = &xpose[w][0];
  if (B == 4) {
    seg_tile<4>(qseg, kseg, out, buf, cpred, 0, Q, K, q0, k0, lane, h, ln);
  } else {
    for (int b0 = 0; b0 < B; ++b0)
      seg_tile<1>(qseg, kseg, out, buf, cpred, b0, Q, K, q0, k0, lane, h, ln);
  }
}

extern "C" void kernel_launch(void* const* d_in, const int* in_sizes, int n_in,
                              void* d_out, int out_size, void* d_ws, size_t ws_size,
                              hipStream_t stream) {
  const int* expl = (const int*)d_in[0];
  const int* qseg = (const int*)d_in[1];
  const int* kseg = (const int*)d_in[2];
  const int* offp = (const int*)d_in[5];
  const int* winp = (const int*)d_in[6];
  float* out = (float*)d_out;

  const long long qk = (long long)in_sizes[0];
  const int Q = (int)llround(sqrt((double)qk));
  const int K = (int)(qk / Q);
  const int B = in_sizes[1] / Q;

  const int tilesQ = Q >> 4;
  const int tilesK = K >> 4;
  const int totalTiles = tilesQ * tilesK;
  const int blocks = (totalTiles + 7) / 8;

  attn_mask_kernel<<<blocks, 256, 0, stream>>>(expl, qseg, kseg, offp, winp,
                                               out, B, Q, K, tilesK, totalTiles);
}
